// Self_Attention_head_43825846288848
// MI455X (gfx1250) — matrix operand order repaired and checked by reading, compiled, not run on hardware
//
#include <hip/hip_runtime.h>
#include <hip/hip_bf16.h>

#define EMB   512
#define HD    64
#define BATCH 16
#define SEQ   2048

typedef __attribute__((ext_vector_type(8)))  float  v8f;
typedef __attribute__((ext_vector_type(16))) __bf16 v16bf;

union Frag { uint4 q[2]; v16bf v; };

__device__ __forceinline__ unsigned short f2bf(float f) {
    union { __bf16 h; unsigned short s; } u;
    u.h = (__bf16)f;
    return u.s;
}

// ---------------------------------------------------------------------------
// Kernel 0: W [512x64] f32 -> Wt [3][64][512] bf16 (transposed, so projection
// B-fragments become contiguous 32B loads)
// ---------------------------------------------------------------------------
__global__ void wt_kernel(const float* __restrict__ Wq,
                          const float* __restrict__ Wk,
                          const float* __restrict__ Wv,
                          unsigned short* __restrict__ Wt) {
    int idx = blockIdx.x * blockDim.x + threadIdx.x;
    if (idx >= 3 * HD * EMB) return;
    int m   = idx / (HD * EMB);
    int rem = idx % (HD * EMB);
    int n   = rem / EMB;
    int k   = rem % EMB;
    const float* W = (m == 0) ? Wq : (m == 1) ? Wk : Wv;
    Wt[idx] = f2bf(W[k * HD + n]);
}

// ---------------------------------------------------------------------------
// Kernel 1: QKV projection. One wave per 16-row tile of x.
//  - x tile (16x512 f32) -> bf16 in LDS (pitch-padded, conflict-free b128 reads)
//  - 3 matrices x 4 n-tiles x 16 K-steps of v_wmma_f32_16x16x32_bf16
//  - Q,K stored [B,S,64] bf16 ; V stored transposed [B,64,S] bf16
// ---------------------------------------------------------------------------
#define XPITCH 520   // 512 + 8 bf16 pad per row -> rotates LDS banks per row

__global__ void __launch_bounds__(32)
qkv_kernel(const float* __restrict__ x,
           const unsigned short* __restrict__ Wt,   // [3][64][512] bf16
           unsigned short* __restrict__ Q,
           unsigned short* __restrict__ K,
           unsigned short* __restrict__ Vt) {
    __shared__ unsigned short xt[16 * XPITCH];

    const int lane = threadIdx.x;        // 0..31
    const int wg   = blockIdx.x;         // 0..B*S/16-1
    const int b    = wg >> 7;            // / (SEQ/16)
    const int row0 = (wg & 127) << 4;
    const int col  = lane & 15;
    const int half = lane >> 4;

    // ---- stage x tile as bf16 in LDS ----
    const float* xb = x + (size_t)(b * SEQ + row0) * EMB;
    for (int i = lane; i < 16 * 128; i += 32) {
        int r  = i >> 7;
        int c4 = i & 127;
        float4 f = *(const float4*)(xb + r * EMB + c4 * 4);
        uint2 pk;
        pk.x = (unsigned)f2bf(f.x) | ((unsigned)f2bf(f.y) << 16);
        pk.y = (unsigned)f2bf(f.z) | ((unsigned)f2bf(f.w) << 16);
        *(uint2*)(&xt[r * XPITCH + c4 * 4]) = pk;
    }
    __syncthreads();

    for (int m = 0; m < 3; ++m) {
        const unsigned short* W = Wt + m * (HD * EMB);
        for (int nt = 0; nt < 4; ++nt) {
            const int ncol = nt * 16 + col;
            v8f acc = {};
            for (int ks = 0; ks < 16; ++ks) {
                // A: row = lane%16, 16 contiguous K at 16h
                Frag a;
                const unsigned short* ap = &xt[col * XPITCH + ks * 32 + half * 16];
                a.q[0] = *(const uint4*)(ap);
                a.q[1] = *(const uint4*)(ap + 8);
                // B: col = lane%16, 16 contiguous K at 16h
                Frag bf;
                const unsigned short* bp = W + (size_t)ncol * EMB + ks * 32 + half * 16;
                bf.q[0] = *(const uint4*)(bp);
                bf.q[1] = *(const uint4*)(bp + 8);
                acc = __builtin_amdgcn_wmma_f32_16x16x32_bf16(
                        false, a.v, false, bf.v, (short)0, acc, false, false);
            }
            // C layout: M = r + 8*half, N = lane%16
            if (m < 2) {
                unsigned short* dst = (m == 0) ? Q : K;
#pragma unroll
                for (int r = 0; r < 8; ++r) {
                    int M = r + half * 8;
                    dst[(size_t)(b * SEQ + row0 + M) * HD + ncol] = f2bf(acc[r]);
                }
            } else {
#pragma unroll
                for (int r = 0; r < 8; ++r) {
                    int M = r + half * 8;
                    Vt[(size_t)(b * HD + ncol) * SEQ + row0 + M] = f2bf(acc[r]);
                }
            }
        }
    }
}

// ---------------------------------------------------------------------------
// Kernel 2: causal flash attention. One wave per (batch, 16-query tile).
// Key blocks of 32; online softmax with cross-lane reductions matching the
// WMMA C layout; P converted C-layout -> A-layout through LDS.
// ---------------------------------------------------------------------------
#define PPITCH 40    // 32 + 8 bf16 pad -> conflict-free b128 A-frag reads

__global__ void __launch_bounds__(32)
attn_kernel(const unsigned short* __restrict__ Q,
            const unsigned short* __restrict__ K,
            const unsigned short* __restrict__ Vt,
            float* __restrict__ out) {
    __shared__ unsigned short pt[16 * PPITCH];

    const int lane = threadIdx.x;
    const int wg   = blockIdx.x;
    const int b    = wg >> 7;
    const int m0   = (wg & 127) << 4;
    const int col  = lane & 15;
    const int half = lane >> 4;

    // Q as A-fragments, kept in registers (row = lane%16)
    Frag qa[2];
    {
        const unsigned short* qp = Q + (size_t)(b * SEQ + m0 + col) * HD;
#pragma unroll
        for (int f = 0; f < 2; ++f) {
            qa[f].q[0] = *(const uint4*)(qp + f * 32 + half * 16);
            qa[f].q[1] = *(const uint4*)(qp + f * 32 + half * 16 + 8);
        }
    }

    float mrow[8], lrow[8];
    v8f o[4];
#pragma unroll
    for (int r = 0; r < 8; ++r) { mrow[r] = -1e30f; lrow[r] = 0.f; }
#pragma unroll
    for (int nt = 0; nt < 4; ++nt) o[nt] = (v8f){};

    for (int kb0 = 0; kb0 < m0 + 16; kb0 += 32) {
        // ---- scores S = Q * K^T for 32 keys (two 16-key C tiles) ----
        v8f s0 = {}, s1 = {};
#pragma unroll
        for (int f = 0; f < 2; ++f) {   // head-dim K chunks 0-31, 32-63
            Frag kb0f, kb1f;
            const unsigned short* kp0 =
                K + (size_t)(b * SEQ + kb0 + col) * HD + f * 32 + half * 16;
            const unsigned short* kp1 =
                K + (size_t)(b * SEQ + kb0 + 16 + col) * HD + f * 32 + half * 16;
            kb0f.q[0] = *(const uint4*)(kp0);
            kb0f.q[1] = *(const uint4*)(kp0 + 8);
            kb1f.q[0] = *(const uint4*)(kp1);
            kb1f.q[1] = *(const uint4*)(kp1 + 8);
            s0 = __builtin_amdgcn_wmma_f32_16x16x32_bf16(
                    false, qa[f].v, false, kb0f.v, (short)0, s0, false, false);
            s1 = __builtin_amdgcn_wmma_f32_16x16x32_bf16(
                    false, qa[f].v, false, kb1f.v, (short)0, s1, false, false);
        }

        // ---- scale 1/sqrt(64), causal mask, online softmax update ----
        float p0[8], p1[8];
#pragma unroll
        for (int r = 0; r < 8; ++r) {
            int qrow = m0 + r + half * 8;
            float a0 = (kb0 + col      <= qrow) ? s0[r] * 0.125f : -1e30f;
            float a1 = (kb0 + 16 + col <= qrow) ? s1[r] * 0.125f : -1e30f;
            float mx = fmaxf(a0, a1);
#pragma unroll
            for (int off = 1; off < 16; off <<= 1)      // stays inside 16-lane half
                mx = fmaxf(mx, __shfl_xor(mx, off, 32));
            float mnew  = fmaxf(mrow[r], mx);
            float alpha = __expf(mrow[r] - mnew);
            float e0 = __expf(a0 - mnew);
            float e1 = __expf(a1 - mnew);
            float rs = e0 + e1;
#pragma unroll
            for (int off = 1; off < 16; off <<= 1)
                rs += __shfl_xor(rs, off, 32);
            lrow[r] = lrow[r] * alpha + rs;
            mrow[r] = mnew;
            p0[r] = e0;
            p1[r] = e1;
#pragma unroll
            for (int nt = 0; nt < 4; ++nt) o[nt][r] *= alpha;
        }

        // ---- P: C-layout -> row-major [16][32] bf16 in LDS ----
#pragma unroll
        for (int r = 0; r < 8; ++r) {
            int M = r + half * 8;
            pt[M * PPITCH + col]      = f2bf(p0[r]);
            pt[M * PPITCH + 16 + col] = f2bf(p1[r]);
        }
        __syncthreads();

        // ---- P as A-fragment, V as B-fragments; O += P*V ----
        Frag pa;
        {
            const unsigned short* pp = &pt[col * PPITCH + half * 16];
            pa.q[0] = *(const uint4*)(pp);
            pa.q[1] = *(const uint4*)(pp + 8);
        }
#pragma unroll
        for (int nt = 0; nt < 4; ++nt) {
            Frag vb;
            const unsigned short* vp =
                Vt + (size_t)(b * HD + nt * 16 + col) * SEQ + kb0 + half * 16;
            vb.q[0] = *(const uint4*)(vp);
            vb.q[1] = *(const uint4*)(vp + 8);
            o[nt] = __builtin_amdgcn_wmma_f32_16x16x32_bf16(
                        false, pa.v, false, vb.v, (short)0, o[nt], false, false);
        }
        __syncthreads();
    }

    // ---- epilogue: divide by row sum, store fp32 ----
#pragma unroll
    for (int nt = 0; nt < 4; ++nt) {
#pragma unroll
        for (int r = 0; r < 8; ++r) {
            int M = r + half * 8;
            out[(size_t)(b * SEQ + m0 + M) * HD + nt * 16 + col] = o[nt][r] / lrow[r];
        }
    }
}

// ---------------------------------------------------------------------------
extern "C" void kernel_launch(void* const* d_in, const int* in_sizes, int n_in,
                              void* d_out, int out_size, void* d_ws, size_t ws_size,
                              hipStream_t stream) {
    (void)in_sizes; (void)n_in; (void)out_size; (void)ws_size;
    const float* x  = (const float*)d_in[0];
    // d_in[1] = mask (causal tril) -> computed analytically, not read
    const float* Wq = (const float*)d_in[2];
    const float* Wk = (const float*)d_in[3];
    const float* Wv = (const float*)d_in[4];

    unsigned short* Wt = (unsigned short*)d_ws;                        // 192 KB
    unsigned short* Qw = (unsigned short*)((char*)d_ws + (1u  << 20)); // 4 MB
    unsigned short* Kw = (unsigned short*)((char*)d_ws + (5u  << 20)); // 4 MB
    unsigned short* Vt = (unsigned short*)((char*)d_ws + (9u  << 20)); // 4 MB

    wt_kernel<<<(3 * HD * EMB + 255) / 256, 256, 0, stream>>>(Wq, Wk, Wv, Wt);
    qkv_kernel<<<BATCH * SEQ / 16, 32, 0, stream>>>(x, Wt, Qw, Kw, Vt);
    attn_kernel<<<BATCH * SEQ / 16, 32, 0, stream>>>(Qw, Kw, Vt, (float*)d_out);
}
